// TransformerLayer_38027640439454
// MI455X (gfx1250) — matrix operand order repaired and checked by reading, not compiled, not run on hardware
//
#include <hip/hip_runtime.h>
#include <hip/hip_bf16.h>

#define HID 1024
#define HEADS 16
#define HDIM 64
#define BATCH 4
#define SEQ 2048
#define MTOT (BATCH * SEQ)       // 8192
#define FFD (HID * 4)            // 4096

typedef __attribute__((ext_vector_type(16))) __bf16 v16bf;
typedef __attribute__((ext_vector_type(8)))  __bf16 v8bf;
typedef __attribute__((ext_vector_type(8)))  float  v8f;
typedef __attribute__((ext_vector_type(4)))  unsigned int u32x4;
typedef __attribute__((ext_vector_type(8)))  int i32x8;
typedef __attribute__((ext_vector_type(4)))  int i32x4;

// ---------------------------------------------------------------------------
// Fragment loaders (wave32 CDNA5 WMMA layouts)
// A-matrix 16x32 bf16: lane<16 -> row=lane,    K = {0..7, 16..23}
//                      lane>=16 -> row=lane-16, K = {8..15, 24..31}
// B-matrix 32x16 bf16: lane<16 -> col=lane,    K = {0..7, 16..23}
//                      lane>=16 -> col=lane-16, K = {8..15, 24..31}
// ---------------------------------------------------------------------------
__device__ __forceinline__ v16bf ld_frag_a(const __bf16* rowp, int lh) {
    v8bf lo = *(const v8bf*)(rowp + lh * 8);
    v8bf hi = *(const v8bf*)(rowp + 16 + lh * 8);
    v16bf f;
#pragma unroll
    for (int i = 0; i < 8; ++i) { f[i] = lo[i]; f[i + 8] = hi[i]; }
    return f;
}

__device__ __forceinline__ v16bf ld_frag_b(const __bf16* rowp, int lh) {
    v8bf lo = *(const v8bf*)(rowp + lh * 8);
    v8bf hi = *(const v8bf*)(rowp + 16 + lh * 8);
    v16bf f;
#pragma unroll
    for (int i = 0; i < 8; ++i) { f[i] = lo[i]; f[i + 8] = hi[i]; }
    return f;
}

__device__ __forceinline__ v8f wmma_bf16(v16bf a, v16bf b, v8f c) {
    return __builtin_amdgcn_wmma_f32_16x16x32_bf16(
        false, a, false, b, (short)0, c, false, false);
}

// ---------------------------------------------------------------------------
// Tensor Data Mover: 2D tile (tile_w x tile_h, 2-byte elems) Global -> LDS,
// LDS padding of 4 DWORDs after every 16 DWORDs (row 32 bf16 -> stride 40).
// ---------------------------------------------------------------------------
__device__ __forceinline__ void tdm_load_2d_bf16(unsigned lds_addr,
                                                 unsigned long long gaddr,
                                                 unsigned tensor_w, unsigned tensor_h,
                                                 unsigned tile_w, unsigned tile_h,
                                                 unsigned long long row_stride) {
    u32x4 g0;
    g0[0] = 1u;                                      // count=1, user mode
    g0[1] = lds_addr;                                // lds_addr[31:0]
    g0[2] = (unsigned)gaddr;                         // global_addr[31:0]
    g0[3] = (unsigned)(gaddr >> 32) | (2u << 30);    // global_addr[56:32] | type=2
    i32x8 g1;
    g1[0] = (int)((1u << 16)        // data_size = 2 bytes
                 | (1u << 20)       // pad_enable
                 | (3u << 22)       // pad_interval: 16 DWORDs
                 | (3u << 25));     // pad_amount:   4 DWORDs
    g1[1] = (int)((tensor_w & 0xFFFFu) << 16);                       // dim0[15:0]
    g1[2] = (int)(((tensor_w >> 16) & 0xFFFFu) | ((tensor_h & 0xFFFFu) << 16));
    g1[3] = (int)(((tensor_h >> 16) & 0xFFFFu) | ((tile_w & 0xFFFFu) << 16));
    g1[4] = (int)(tile_h & 0xFFFFu);                                 // tile_dim1
    g1[5] = (int)(unsigned)(row_stride & 0xFFFFFFFFu);               // dim0_stride lo
    g1[6] = (int)(unsigned)((row_stride >> 32) & 0xFFFFu);           // dim0_stride hi
    g1[7] = 0;
    i32x4 z4 = {};
#if __clang_major__ >= 23
    i32x8 z8 = {};
    __builtin_amdgcn_tensor_load_to_lds(g0, g1, z4, z4, z8, 0);
#else
    __builtin_amdgcn_tensor_load_to_lds(g0, g1, z4, z4, 0);
#endif
}

// ---------------------------------------------------------------------------
// fp32 -> bf16 convert (weights)
// ---------------------------------------------------------------------------
__global__ void cvt_bf16_kernel(const float* __restrict__ in,
                                __bf16* __restrict__ out, int n) {
    int i = blockIdx.x * 256 + threadIdx.x;
    if (i < n) out[i] = (__bf16)in[i];
}

// ---------------------------------------------------------------------------
// LayerNorm: one block per row (1024 elems, 256 threads x 4), bf16 output
// ---------------------------------------------------------------------------
__global__ __launch_bounds__(256)
void ln_bf16_kernel(const float* __restrict__ x, const float* __restrict__ g,
                    const float* __restrict__ bta, __bf16* __restrict__ out) {
    const int row = blockIdx.x;
    const int tid = threadIdx.x;
    const int lane = tid & 31, w = tid >> 5;
    const float* xr = x + (size_t)row * HID;
    float4 v = *(const float4*)(xr + tid * 4);
    float s = v.x + v.y + v.z + v.w;
    float s2 = v.x * v.x + v.y * v.y + v.z * v.z + v.w * v.w;
#pragma unroll
    for (int o = 16; o > 0; o >>= 1) {
        s += __shfl_down(s, o, 32);
        s2 += __shfl_down(s2, o, 32);
    }
    __shared__ float rs[8], rs2[8], stat[2];
    if (lane == 0) { rs[w] = s; rs2[w] = s2; }
    __syncthreads();
    if (tid == 0) {
        float a = 0.f, b2 = 0.f;
#pragma unroll
        for (int i = 0; i < 8; ++i) { a += rs[i]; b2 += rs2[i]; }
        float mu = a * (1.0f / HID);
        float var = b2 * (1.0f / HID) - mu * mu;
        stat[0] = mu;
        stat[1] = rsqrtf(var + 1e-5f);
    }
    __syncthreads();
    const float mu = stat[0], rstd = stat[1];
    const float fv[4] = {v.x, v.y, v.z, v.w};
#pragma unroll
    for (int j = 0; j < 4; ++j) {
        int c = tid * 4 + j;
        out[(size_t)row * HID + c] = (__bf16)((fv[j] - mu) * rstd * g[c] + bta[c]);
    }
}

// ---------------------------------------------------------------------------
// Tiled WMMA GEMM:  out[M,N] = A[M,K] @ W[N,K]^T + bias  (+ epilogue)
// Block: 256 thr = 8 waves (4x2), wave computes 32x64 (2x4 WMMA tiles).
// Block tile 128x128, K-step 32. A tile via cooperative b128 loads;
// B (weight) tile via the Tensor Data Mover straight into padded LDS.
// EPI 0: bf16 out; EPI 1: bf16 relu out; EPI 2: f32 out = res + acc + bias
// ---------------------------------------------------------------------------
#define LDT 40  // padded bf16 row stride in LDS (32 data + 8 pad)

template <int EPI>
__global__ __launch_bounds__(256)
void gemm_bf16_wmma(const __bf16* __restrict__ A, const __bf16* __restrict__ W,
                    const float* __restrict__ bias, const float* __restrict__ res,
                    float* __restrict__ outf, __bf16* __restrict__ outb,
                    int M, int N, int K) {
    __shared__ __bf16 As[128 * LDT];
    __shared__ __bf16 Bs[128 * LDT];
    const int tid = threadIdx.x;
    const int lane = tid & 31;
    const int wv = tid >> 5;
    const int wm = wv >> 1, wn = wv & 1;   // 4 x 2 wave grid
    const int lr = lane & 15, lh = lane >> 4;
    const int m0 = blockIdx.y * 128, n0 = blockIdx.x * 128;
    const int arow = tid >> 2, ac = (tid & 3) * 8;
    const unsigned bs_lds = (unsigned)(size_t)(&Bs[0]);

    v8f acc[2][4] = {};
    for (int k0 = 0; k0 < K; k0 += 32) {
        uint4 ga0 = *(const uint4*)(A + (size_t)(m0 + arow) * K + k0 + ac);
        uint4 ga1 = *(const uint4*)(A + (size_t)(m0 + 64 + arow) * K + k0 + ac);
        __syncthreads();   // previous iteration's fragment reads complete
        if (tid < 32) {    // wave 0 drives the TDM for the 128x32 weight tile
            tdm_load_2d_bf16(bs_lds,
                             (unsigned long long)(size_t)(W + (size_t)n0 * K + k0),
                             (unsigned)K, (unsigned)N,
                             32u, 128u, (unsigned long long)K);
        }
        *(uint4*)(As + arow * LDT + ac) = ga0;
        *(uint4*)(As + (64 + arow) * LDT + ac) = ga1;
        if (tid < 32) __builtin_amdgcn_s_wait_tensorcnt(0);
        __syncthreads();

        v16bf af[2], bfm[4];
#pragma unroll
        for (int i = 0; i < 2; ++i)
            af[i] = ld_frag_a(As + (wm * 32 + i * 16 + lr) * LDT, lh);
#pragma unroll
        for (int i = 0; i < 4; ++i)
            bfm[i] = ld_frag_b(Bs + (wn * 64 + i * 16 + lr) * LDT, lh);
#pragma unroll
        for (int mi = 0; mi < 2; ++mi)
#pragma unroll
            for (int ni = 0; ni < 4; ++ni)
                acc[mi][ni] = wmma_bf16(af[mi], bfm[ni], acc[mi][ni]);
    }

#pragma unroll
    for (int mi = 0; mi < 2; ++mi)
#pragma unroll
        for (int ni = 0; ni < 4; ++ni) {
            const int colg = n0 + wn * 64 + ni * 16 + lr;
            const float bb = bias[colg];
#pragma unroll
            for (int r = 0; r < 8; ++r) {
                const int rowg = m0 + wm * 32 + mi * 16 + lh * 8 + r;
                const size_t idx = (size_t)rowg * N + colg;
                float vv = acc[mi][ni][r] + bb;
                if (EPI == 0)      outb[idx] = (__bf16)vv;
                else if (EPI == 1) outb[idx] = (__bf16)fmaxf(vv, 0.f);
                else               outf[idx] = res[idx] + vv;
            }
        }
}

// ---------------------------------------------------------------------------
// V transpose: vt[b,h,d,s] = v[b,s,h*64+d]  (bf16)
// ---------------------------------------------------------------------------
__global__ void vtrans_kernel(const __bf16* __restrict__ v, __bf16* __restrict__ vt) {
    size_t i = (size_t)blockIdx.x * 256 + threadIdx.x;
    int s = (int)(i & (SEQ - 1));
    size_t t = i >> 11;            // / SEQ
    int d = (int)(t & 63); t >>= 6;
    int h = (int)(t & 15); int b = (int)(t >> 4);
    vt[i] = v[((size_t)b * SEQ + s) * HID + h * HDIM + d];
}

// ---------------------------------------------------------------------------
// Flash-style attention. One wave per (b, h, 16-query tile).
// Score tiles computed TRANSPOSED (M=key, N=query) so the f32 C/D layout of
// exp(scores) maps directly onto the bf16 A-fragment layout of P for P@V.
// ---------------------------------------------------------------------------
__global__ __launch_bounds__(256)
void attn_wmma_kernel(const __bf16* __restrict__ qb, const __bf16* __restrict__ kb,
                      const __bf16* __restrict__ vt, __bf16* __restrict__ out) {
    const int lane = threadIdx.x & 31;
    const int wid = blockIdx.x * 8 + (threadIdx.x >> 5);
    const int qt = wid & (SEQ / 16 - 1);
    const int h = (wid >> 7) & (HEADS - 1);
    const int b = wid >> 11;
    const int lr = lane & 15, lh = lane >> 4;
    const int q0 = qt * 16;

    // Q as B-fragments (dims 0..31, 32..63)
    const __bf16* qrow = qb + ((size_t)b * SEQ + q0 + lr) * HID + h * HDIM;
    v16bf qf[2];
    qf[0] = ld_frag_b(qrow, lh);
    qf[1] = ld_frag_b(qrow + 32, lh);

    const __bf16* kbase = kb + (size_t)b * SEQ * HID + h * HDIM;
    const __bf16* vbase = vt + ((size_t)(b * HEADS + h) * HDIM) * SEQ;

    v8f oacc[4] = {};
    float m_run = -3.0e38f, l_run = 0.f;

    for (int k0 = 0; k0 < SEQ; k0 += 32) {
        // prefetch next key tile rows into cache (global_prefetch_b8)
        __builtin_prefetch(kbase + (size_t)(k0 + 32 + lr) * HID, 0, 1);
        // scores^T for keys k0..k0+15 (s0) and k0+16..k0+31 (s1)
        v8f s0 = {}, s1 = {};
        {
            const __bf16* kr0 = kbase + (size_t)(k0 + lr) * HID;
            s0 = wmma_bf16(ld_frag_a(kr0, lh), qf[0], s0);
            s0 = wmma_bf16(ld_frag_a(kr0 + 32, lh), qf[1], s0);
            const __bf16* kr1 = kbase + (size_t)(k0 + 16 + lr) * HID;
            s1 = wmma_bf16(ld_frag_a(kr1, lh), qf[0], s1);
            s1 = wmma_bf16(ld_frag_a(kr1 + 32, lh), qf[1], s1);
        }
        float mloc = -3.0e38f;
        float f0[8], f1[8];
#pragma unroll
        for (int r = 0; r < 8; ++r) {
            f0[r] = s0[r] * 0.125f;
            f1[r] = s1[r] * 0.125f;
            mloc = fmaxf(mloc, fmaxf(f0[r], f1[r]));
        }
        mloc = fmaxf(mloc, __shfl_xor(mloc, 16, 32));   // combine wave halves
        const float mnew = fmaxf(m_run, mloc);
        const float alpha = __expf(m_run - mnew);
        float ps = 0.f;
        float e0[8], e1[8];
#pragma unroll
        for (int r = 0; r < 8; ++r) {
            e0[r] = __expf(f0[r] - mnew);
            e1[r] = __expf(f1[r] - mnew);
            ps += e0[r] + e1[r];
        }
        ps += __shfl_xor(ps, 16, 32);
        l_run = l_run * alpha + ps;
        m_run = mnew;

        // P (16 queries x 32 keys) A-fragment: direct from score C-layout
        v16bf pf;
#pragma unroll
        for (int i = 0; i < 8; ++i) {
            pf[i] = (__bf16)e0[i];
            pf[i + 8] = (__bf16)e1[i];
        }

        // per-output-row rescale factors (query = r + 8*lh lives in lane q)
        float al[8];
#pragma unroll
        for (int r = 0; r < 8; ++r) al[r] = __shfl(alpha, r + 8 * lh, 32);

#pragma unroll
        for (int nc = 0; nc < 4; ++nc) {
#pragma unroll
            for (int r = 0; r < 8; ++r) oacc[nc][r] *= al[r];
            const __bf16* vr = vbase + (size_t)(nc * 16 + lr) * SEQ + k0;
            oacc[nc] = wmma_bf16(pf, ld_frag_b(vr, lh), oacc[nc]);
        }
    }

    const float inv = 1.0f / l_run;
    float il[8];
#pragma unroll
    for (int r = 0; r < 8; ++r) il[r] = __shfl(inv, r + 8 * lh, 32);
#pragma unroll
    for (int nc = 0; nc < 4; ++nc)
#pragma unroll
        for (int r = 0; r < 8; ++r) {
            const int qq = q0 + 8 * lh + r;
            const int d = nc * 16 + lr;
            out[((size_t)b * SEQ + qq) * HID + h * HDIM + d] =
                (__bf16)(oacc[nc][r] * il[r]);
        }
}

// ---------------------------------------------------------------------------
extern "C" void kernel_launch(void* const* d_in, const int* in_sizes, int n_in,
                              void* d_out, int out_size, void* d_ws, size_t ws_size,
                              hipStream_t stream) {
    const float* x = (const float*)d_in[0];
    const float* n1g = (const float*)d_in[1];
    const float* n1b = (const float*)d_in[2];
    const float* wq = (const float*)d_in[3];
    const float* bq = (const float*)d_in[4];
    const float* wk = (const float*)d_in[5];
    const float* bk = (const float*)d_in[6];
    const float* wv = (const float*)d_in[7];
    const float* bv = (const float*)d_in[8];
    const float* wo = (const float*)d_in[9];
    const float* bo = (const float*)d_in[10];
    const float* n2g = (const float*)d_in[11];
    const float* n2b = (const float*)d_in[12];
    const float* w1 = (const float*)d_in[13];
    const float* b1 = (const float*)d_in[14];
    const float* w2 = (const float*)d_in[15];
    const float* b2 = (const float*)d_in[16];
    float* outp = (float*)d_out;

    char* wsb = (char*)d_ws;
    size_t off = 0;
    auto alloc = [&](size_t bytes) -> void* {
        void* p = wsb + off;
        off = (off + bytes + 255) & ~(size_t)255;
        return p;
    };
    __bf16* wqb = (__bf16*)alloc((size_t)HID * HID * 2);
    __bf16* wkb = (__bf16*)alloc((size_t)HID * HID * 2);
    __bf16* wvb = (__bf16*)alloc((size_t)HID * HID * 2);
    __bf16* wob = (__bf16*)alloc((size_t)HID * HID * 2);
    __bf16* w1b = (__bf16*)alloc((size_t)FFD * HID * 2);
    __bf16* w2b = (__bf16*)alloc((size_t)HID * FFD * 2);
    __bf16* h1 = (__bf16*)alloc((size_t)MTOT * HID * 2);
    __bf16* qb = (__bf16*)alloc((size_t)MTOT * HID * 2);
    __bf16* kbuf = (__bf16*)alloc((size_t)MTOT * HID * 2);
    __bf16* vbuf = (__bf16*)alloc((size_t)MTOT * HID * 2);
    __bf16* vtb = (__bf16*)alloc((size_t)MTOT * HID * 2);
    __bf16* aout = (__bf16*)alloc((size_t)MTOT * HID * 2);
    float* x1 = (float*)alloc((size_t)MTOT * HID * 4);
    __bf16* h2 = (__bf16*)alloc((size_t)MTOT * HID * 2);
    __bf16* ffh = (__bf16*)alloc((size_t)MTOT * FFD * 2);
    (void)ws_size; (void)in_sizes; (void)n_in; (void)out_size;

    // weight conversions
    const int nw = HID * HID;
    cvt_bf16_kernel<<<(nw + 255) / 256, 256, 0, stream>>>(wq, wqb, nw);
    cvt_bf16_kernel<<<(nw + 255) / 256, 256, 0, stream>>>(wk, wkb, nw);
    cvt_bf16_kernel<<<(nw + 255) / 256, 256, 0, stream>>>(wv, wvb, nw);
    cvt_bf16_kernel<<<(nw + 255) / 256, 256, 0, stream>>>(wo, wob, nw);
    const int nw1 = FFD * HID;
    cvt_bf16_kernel<<<(nw1 + 255) / 256, 256, 0, stream>>>(w1, w1b, nw1);
    cvt_bf16_kernel<<<(nw1 + 255) / 256, 256, 0, stream>>>(w2, w2b, nw1);

    // LN1
    ln_bf16_kernel<<<MTOT, 256, 0, stream>>>(x, n1g, n1b, h1);

    // QKV projections (block tile 128x128)
    dim3 gq(HID / 128, MTOT / 128);
    gemm_bf16_wmma<0><<<gq, 256, 0, stream>>>(h1, wqb, bq, nullptr, nullptr, qb, MTOT, HID, HID);
    gemm_bf16_wmma<0><<<gq, 256, 0, stream>>>(h1, wkb, bk, nullptr, nullptr, kbuf, MTOT, HID, HID);
    gemm_bf16_wmma<0><<<gq, 256, 0, stream>>>(h1, wvb, bv, nullptr, nullptr, vbuf, MTOT, HID, HID);

    // V transpose to [B,H,D,S]
    vtrans_kernel<<<(int)(((size_t)MTOT * HID) / 256), 256, 0, stream>>>(vbuf, vtb);

    // attention: B*H*(S/16) = 8192 waves -> 1024 blocks of 8 waves
    attn_wmma_kernel<<<(BATCH * HEADS * (SEQ / 16)) / 8, 256, 0, stream>>>(qb, kbuf, vtb, aout);

    // O projection + residual (f32)
    gemm_bf16_wmma<2><<<gq, 256, 0, stream>>>(aout, wob, bo, x, x1, nullptr, MTOT, HID, HID);

    // LN2
    ln_bf16_kernel<<<MTOT, 256, 0, stream>>>(x1, n2g, n2b, h2);

    // FFN1 (relu) and FFN2 (+ residual -> d_out)
    dim3 gf1(FFD / 128, MTOT / 128);
    gemm_bf16_wmma<1><<<gf1, 256, 0, stream>>>(h2, w1b, b1, nullptr, nullptr, ffh, MTOT, FFD, HID);
    gemm_bf16_wmma<2><<<gq, 256, 0, stream>>>(ffh, w2b, b2, x1, outp, nullptr, MTOT, HID, FFD);
}
